// AIGCN_12979391168623
// MI455X (gfx1250) — matrix operand order repaired and checked by reading, not compiled, not run on hardware
//
#include <hip/hip_runtime.h>

typedef __bf16 bf16;
typedef __attribute__((ext_vector_type(16))) __bf16 v16bf;
typedef __attribute__((ext_vector_type(8)))  float  v8f;

// ---------------- helpers ----------------

__device__ inline v8f vzero8() {
  v8f z;
#pragma unroll
  for (int r = 0; r < 8; ++r) z[r] = 0.f;
  return z;
}

__device__ inline unsigned packf(float lo, float hi) {
  union { unsigned u; bf16 b[2]; } cv;
  cv.b[0] = (bf16)lo; cv.b[1] = (bf16)hi;
  return cv.u;
}

__device__ inline unsigned packh(bf16 lo, bf16 hi) {
  union { unsigned u; bf16 b[2]; } cv;
  cv.b[0] = lo; cv.b[1] = hi;
  return cv.u;
}

union FragCvt { uint4 q[2]; v16bf f; };

// A fragment (16x32 bf16, M x K). LDS: row-major packed K-pairs, strideU % 4 == 0.
// ISA layout: lane(m,h): v0..3 -> K 8h+2v (kp 4h+v), v4..7 -> K 16+8h+2(v-4)
// (kp 8+4h+v-4). Both runs contiguous -> two b128 loads.
__device__ inline v16bf frag_a_lds(const unsigned* p, int strideU) {
  const int lane = threadIdx.x & 31;
  const int m = lane & 15, h = lane >> 4;
  const unsigned* q = p + m * strideU + 4 * h;
  FragCvt cv;
  cv.q[0] = *(const uint4*)q;
  cv.q[1] = *(const uint4*)(q + 8);
  return cv.f;
}

// B fragment (32x16 bf16, K x N). LDS: n-major packed K-pairs: p[n*strideU + kp].
// ISA layout: lane(n,h): v0..3 -> K 8h+2v (kp 4h+v), v4..7 -> K 16+8h+2(v-4) (kp 8+4h+v-4).
__device__ inline v16bf frag_b_lds(const unsigned* p, int strideU, int n0) {
  const int lane = threadIdx.x & 31;
  const int n = n0 + (lane & 15), h = lane >> 4;
  const unsigned* q = p + n * strideU + 4 * h;
  FragCvt cv;
  cv.q[0] = *(const uint4*)q;
  cv.q[1] = *(const uint4*)(q + 8);
  return cv.f;
}

__device__ inline v8f wmma_bf16(v16bf a, v16bf b, v8f c) {
  return __builtin_amdgcn_wmma_f32_16x16x32_bf16(false, a, false, b, (short)0, c,
                                                 false, false);
}

// Row softmax over 64 columns held as 4 accumulators x 16 half-wave lanes.
// C-layout: element r of each v8f is row (8*h + r); column = 16*t + (lane&15).
__device__ inline void row_softmax4(v8f& a0, v8f& a1, v8f& a2, v8f& a3) {
#pragma unroll
  for (int r = 0; r < 8; ++r) {
    float m = fmaxf(fmaxf(a0[r], a1[r]), fmaxf(a2[r], a3[r]));
#pragma unroll
    for (int k = 1; k <= 8; k <<= 1) m = fmaxf(m, __shfl_xor(m, k, 32));
    a0[r] = __expf(a0[r] - m); a1[r] = __expf(a1[r] - m);
    a2[r] = __expf(a2[r] - m); a3[r] = __expf(a3[r] - m);
    float s = a0[r] + a1[r] + a2[r] + a3[r];
#pragma unroll
    for (int k = 1; k <= 8; k <<= 1) s += __shfl_xor(s, k, 32);
    const float inv = 1.f / s;
    a0[r] *= inv; a1[r] *= inv; a2[r] *= inv; a3[r] *= inv;
  }
}

// ---------------- small kernels ----------------

// transpose+convert weights: dst[ib][c][r] = bf16(src[ib][r][c])
__global__ void transcvt_kernel(const float* __restrict__ src, bf16* __restrict__ dst,
                                int batch, int R, int C) {
  const int total = batch * R * C;
  for (int i = blockIdx.x * blockDim.x + threadIdx.x; i < total;
       i += gridDim.x * blockDim.x) {
    const int ib = i / (R * C), rem = i % (R * C);
    const int r = rem / C, c = rem % C;
    dst[(size_t)ib * R * C + (size_t)c * R + r] = (bf16)src[i];
  }
}

__global__ void cvt_kernel(const float* __restrict__ src, bf16* __restrict__ dst, int n) {
  for (int i = blockIdx.x * blockDim.x + threadIdx.x; i < n;
       i += gridDim.x * blockDim.x)
    dst[i] = (bf16)src[i];
}

// proj[b,a] = sum_n occ[b,n]*proj_w[a,n] + proj_b[a]   (block=64 threads, grid=B)
__global__ void proj_kernel(const float* __restrict__ occ, const float* __restrict__ pw,
                            const float* __restrict__ pb, float* __restrict__ proj) {
  const int b = blockIdx.x;
  __shared__ float so[512];
  for (int i = threadIdx.x; i < 512; i += 64) so[i] = occ[(size_t)b * 512 + i];
  __syncthreads();
  const int a = threadIdx.x;
  float s = 0.f;
  for (int n = 0; n < 512; ++n) s += so[n] * pw[(size_t)a * 512 + n];
  proj[b * 64 + a] = s + pb[a];
}

// xm[b][c][n] = x[b][n][c]
__global__ void xpose_kernel(const float* __restrict__ x, float* __restrict__ xm) {
  const int b = blockIdx.x;
  for (int i = threadIdx.x; i < 512 * 64; i += 256) {
    const int n = i >> 6, c = i & 63;
    xm[(size_t)b * 32768 + (size_t)c * 512 + n] = x[(size_t)b * 32768 + i];
  }
}

// one wave per (b,c): m1/m2 gate scalars
__global__ void mgate_kernel(const float* __restrict__ xm, const float* __restrict__ proj,
                             const float* __restrict__ g1w, const float* __restrict__ g2w,
                             const float* __restrict__ g1b, const float* __restrict__ g2b,
                             float* __restrict__ m1buf, float* __restrict__ m2buf) {
  const int wid = (blockIdx.x * blockDim.x + threadIdx.x) >> 5;  // 0..16383
  const int lane = threadIdx.x & 31;
  const int b = wid >> 6, c = wid & 63;
  const float* xr = xm + (size_t)wid * 512;
  float d1 = 0.f, d2 = 0.f, p1 = 0.f, sg2 = 0.f;
#pragma unroll
  for (int i = 0; i < 16; ++i) {
    const float xv = xr[lane + 32 * i];
    d1 += xv * g1w[lane + 32 * i];
    d2 += xv * g2w[lane + 32 * i];
  }
  {
    const float pa = proj[b * 64 + lane], pbv = proj[b * 64 + lane + 32];
    p1 += pa * g1w[512 + lane] + pbv * g1w[512 + lane + 32];
    sg2 += g2w[512 + lane] + g2w[512 + lane + 32];
  }
#pragma unroll
  for (int k = 1; k <= 16; k <<= 1) {
    d1 += __shfl_xor(d1, k, 32); d2 += __shfl_xor(d2, k, 32);
    p1 += __shfl_xor(p1, k, 32); sg2 += __shfl_xor(sg2, k, 32);
  }
  if (lane == 0) {
    m1buf[wid] = fmaxf(0.f, tanhf(d1 + p1 + g1b[0]));
    m2buf[wid] = fmaxf(0.f, tanhf(d2 + proj[b * 64 + c] * sg2 + g2b[0]));
  }
}

// ---------------- WMMA kernels ----------------

// u = xm(64x512) @ llT(512x64) per b; epilogue: relu(m*(u+bias)), row softmax,
// E1[b][c][a] = proj[a]+sm  (which==0)   or   E2[b][a][c] = proj[c]+sm (which==1)
// block = 128 threads (4 waves), grid = 256 (one per b)
__global__ void llsm_kernel(const float* __restrict__ xm, const bf16* __restrict__ llT,
                            const float* __restrict__ llb, const float* __restrict__ mbuf,
                            const float* __restrict__ proj, float* __restrict__ Eout,
                            int which) {
  const int b = blockIdx.x;
  const int rows0 = b * 64;
  const int w = threadIdx.x >> 5, lane = threadIdx.x & 31;
  __shared__ __align__(16) unsigned ldsA[64 * 20];  // [row][kp], 16 kp + pad
  __shared__ __align__(16) unsigned ldsB[64 * 20];  // [n][kp], 16 kp + pad
  __shared__ float sM[64], sP[64], sB[64];
  if (threadIdx.x < 64) {
    sM[threadIdx.x] = mbuf[rows0 + threadIdx.x];
    sP[threadIdx.x] = proj[b * 64 + threadIdx.x];
    sB[threadIdx.x] = llb[threadIdx.x];
  }
  v8f acc[4];
#pragma unroll
  for (int t = 0; t < 4; ++t) acc[t] = vzero8();

  for (int k0 = 0; k0 < 512; k0 += 32) {
    __syncthreads();
    {  // stage A (64 rows x 32 K) from f32, pack pairs
      const int kp = threadIdx.x & 15, rb = threadIdx.x >> 4;  // rb 0..7
#pragma unroll
      for (int i = 0; i < 8; ++i) {
        const int row = rb + 8 * i;
        const float2 f2 =
            *(const float2*)(xm + (size_t)(rows0 + row) * 512 + k0 + 2 * kp);
        ldsA[row * 20 + kp] = packf(f2.x, f2.y);
      }
    }
    {  // stage B (32 K x 64 n) from bf16 k-major -> n-major packed pairs
      const int n = threadIdx.x & 63, kpb = threadIdx.x >> 6;  // 0/1
#pragma unroll
      for (int i = 0; i < 8; ++i) {
        const int kp = kpb + 2 * i;
        const bf16 lo = llT[(size_t)(k0 + 2 * kp) * 64 + n];
        const bf16 hi = llT[(size_t)(k0 + 2 * kp + 1) * 64 + n];
        ldsB[n * 20 + kp] = packh(lo, hi);
      }
    }
    __syncthreads();
    const v16bf a = frag_a_lds(ldsA + (16 * w) * 20, 20);
#pragma unroll
    for (int t = 0; t < 4; ++t) {
      const v16bf bb = frag_b_lds(ldsB, 20, 16 * t);
      acc[t] = wmma_bf16(a, bb, acc[t]);
    }
  }
  __syncthreads();
  const int h = lane >> 4, nl = lane & 15;
#pragma unroll
  for (int t = 0; t < 4; ++t) {
    const float bias = sB[16 * t + nl];
#pragma unroll
    for (int r = 0; r < 8; ++r) {
      const float mval = sM[16 * w + 8 * h + r];
      acc[t][r] = fmaxf(0.f, mval * (acc[t][r] + bias));
    }
  }
  row_softmax4(acc[0], acc[1], acc[2], acc[3]);
#pragma unroll
  for (int t = 0; t < 4; ++t) {
#pragma unroll
    for (int r = 0; r < 8; ++r) {
      const int rowc = 16 * w + 8 * h + r;
      const int col = 16 * t + nl;
      if (which == 0)
        Eout[(size_t)b * 4096 + rowc * 64 + col] = sP[col] + acc[t][r];
      else
        Eout[(size_t)b * 4096 + col * 64 + rowc] = sP[rowc] + acc[t][r];
    }
  }
}

// adj: G = relu(E1(64x64) @ E2(64x64)); row softmax; store adjT[b][c'][c] bf16
// block = 128 (4 waves), grid = 256
__global__ void adj_kernel(const float* __restrict__ E1, const float* __restrict__ E2,
                           bf16* __restrict__ adjT) {
  const int b = blockIdx.x;
  __shared__ __align__(16) unsigned ldsA[64 * 36];  // [row][kp], 32 kp + pad
  __shared__ __align__(16) unsigned ldsB[64 * 36];  // [n][kp], 32 kp + pad
  const int w = threadIdx.x >> 5, lane = threadIdx.x & 31;
  {  // E1 rows c, 32 kpairs each
    const int kp = threadIdx.x & 31, rb = threadIdx.x >> 5;  // 0..3
#pragma unroll
    for (int i = 0; i < 16; ++i) {
      const int row = rb + 4 * i;
      const float2 f2 = *(const float2*)(E1 + (size_t)b * 4096 + row * 64 + 2 * kp);
      ldsA[row * 36 + kp] = packf(f2.x, f2.y);
    }
  }
  {  // E2[a][c'] -> n-major packed pairs
    const int n = threadIdx.x & 63, kpb = threadIdx.x >> 6;  // 0/1
#pragma unroll
    for (int i = 0; i < 16; ++i) {
      const int kp = kpb + 2 * i;
      const float lo = E2[(size_t)b * 4096 + (2 * kp) * 64 + n];
      const float hi = E2[(size_t)b * 4096 + (2 * kp + 1) * 64 + n];
      ldsB[n * 36 + kp] = packf(lo, hi);
    }
  }
  __syncthreads();
  v8f acc[4];
#pragma unroll
  for (int t = 0; t < 4; ++t) acc[t] = vzero8();
#pragma unroll
  for (int ks = 0; ks < 2; ++ks) {
    const v16bf a = frag_a_lds(ldsA + (16 * w) * 36 + 16 * ks, 36);
#pragma unroll
    for (int t = 0; t < 4; ++t) {
      const v16bf bb = frag_b_lds(ldsB + 16 * ks, 36, 16 * t);
      acc[t] = wmma_bf16(a, bb, acc[t]);
    }
  }
#pragma unroll
  for (int t = 0; t < 4; ++t)
#pragma unroll
    for (int r = 0; r < 8; ++r) acc[t][r] = fmaxf(0.f, acc[t][r]);
  row_softmax4(acc[0], acc[1], acc[2], acc[3]);
  const int h = lane >> 4, nl = lane & 15;
#pragma unroll
  for (int t = 0; t < 4; ++t)
#pragma unroll
    for (int r = 0; r < 8; ++r) {
      const int c = 16 * w + 8 * h + r;
      const int cp = 16 * t + nl;
      adjT[(size_t)b * 4096 + cp * 64 + c] = (bf16)acc[t][r];
    }
}

// nconv: out(64 x 128f) = adjT(64x64) @ src(64 x 128f), write into Y seg dstSeg.
// srcIsF32: src = xm f32 (also mirror bf16 into Y seg0 when writeSeg0)
// block = 256 (8 waves), grid = (256, 4)
__global__ void nconv_kernel(const bf16* __restrict__ adjT, const float* __restrict__ xmF,
                             bf16* __restrict__ Y, int srcIsF32, int srcSeg, int dstSeg,
                             int writeSeg0) {
  const int b = blockIdx.x, f0 = blockIdx.y * 128;
  __shared__ __align__(16) unsigned ldsA[64 * 36];    // [row][kp]
  __shared__ __align__(16) unsigned ldsB[128 * 36];   // [n][kp]
  const int tid = threadIdx.x;
  {  // stage adjT (bf16, row-major 64x64) as uints
    const int kp = tid & 31, rb = tid >> 5;  // 0..7
    const unsigned* src = (const unsigned*)(adjT + (size_t)b * 4096);
#pragma unroll
    for (int i = 0; i < 8; ++i) {
      const int row = rb + 8 * i;
      ldsA[row * 36 + kp] = src[row * 32 + kp];
    }
  }
  {  // stage B 64(K) x 128(n), n-major packed pairs
    const int n = tid & 127, kpb = tid >> 7;  // 0/1
#pragma unroll
    for (int i = 0; i < 16; ++i) {
      const int kp = kpb + 2 * i;
      if (srcIsF32) {
        const float lo = xmF[(size_t)(b * 64 + 2 * kp) * 512 + f0 + n];
        const float hi = xmF[(size_t)(b * 64 + 2 * kp + 1) * 512 + f0 + n];
        if (writeSeg0) {
          Y[(size_t)(b * 64 + 2 * kp) * 1536 + f0 + n] = (bf16)lo;
          Y[(size_t)(b * 64 + 2 * kp + 1) * 1536 + f0 + n] = (bf16)hi;
        }
        ldsB[n * 36 + kp] = packf(lo, hi);
      } else {
        const bf16 lo = Y[(size_t)(b * 64 + 2 * kp) * 1536 + srcSeg * 512 + f0 + n];
        const bf16 hi = Y[(size_t)(b * 64 + 2 * kp + 1) * 1536 + srcSeg * 512 + f0 + n];
        ldsB[n * 36 + kp] = packh(lo, hi);
      }
    }
  }
  __syncthreads();
  const int w = tid >> 5, lane = tid & 31;
  const int mr = w >> 1, wc = w & 1;
  v8f acc[4];
#pragma unroll
  for (int t = 0; t < 4; ++t) acc[t] = vzero8();
#pragma unroll
  for (int ks = 0; ks < 2; ++ks) {
    const v16bf a = frag_a_lds(ldsA + (16 * mr) * 36 + 16 * ks, 36);
#pragma unroll
    for (int t = 0; t < 4; ++t) {
      const v16bf bb = frag_b_lds(ldsB + 16 * ks, 36, 64 * wc + 16 * t);
      acc[t] = wmma_bf16(a, bb, acc[t]);
    }
  }
  const int h = lane >> 4, nl = lane & 15;
#pragma unroll
  for (int t = 0; t < 4; ++t)
#pragma unroll
    for (int r = 0; r < 8; ++r) {
      const int c = 16 * mr + 8 * h + r;
      const int n = 64 * wc + 16 * t + nl;
      Y[(size_t)(b * 64 + c) * 1536 + dstSeg * 512 + f0 + n] = (bf16)acc[t][r];
    }
}

// generic GEMM: out[m][n] = epi( A(bf16,[M][lda]) @ Bt(bf16,[K][ldb]) + bias )
// block = 256 (8 waves), tile 128x128, grid = (M/128, N/128)
__global__ void gemm_kernel(const bf16* __restrict__ A, int lda,
                            const bf16* __restrict__ Bt, int ldb,
                            const float* __restrict__ bias, float* __restrict__ outF,
                            bf16* __restrict__ outH, const float* __restrict__ res,
                            int ldo, int K, int relu) {
  const int m0 = blockIdx.x * 128, n0 = blockIdx.y * 128;
  __shared__ __align__(16) unsigned ldsA[128 * 20];  // [row][kp]
  __shared__ __align__(16) unsigned ldsB[128 * 20];  // [n][kp]
  const int tid = threadIdx.x, w = tid >> 5, lane = tid & 31;
  const int wr = w >> 1, wc = w & 1;
  v8f acc[2][4];
#pragma unroll
  for (int i = 0; i < 2; ++i)
#pragma unroll
    for (int t = 0; t < 4; ++t) acc[i][t] = vzero8();

  for (int k0 = 0; k0 < K; k0 += 32) {
    __syncthreads();
    {  // A: 128 rows x 16 kpairs (bf16 global -> uint)
      const int kp = tid & 15, rb = tid >> 4;  // 0..15
#pragma unroll
      for (int i = 0; i < 8; ++i) {
        const int row = rb + 16 * i;
        ldsA[row * 20 + kp] =
            *(const unsigned*)(A + (size_t)(m0 + row) * lda + k0 + 2 * kp);
      }
    }
    {  // B: 16 kpairs x 128 n -> n-major packed pairs
      const int n = tid & 127, kpb = tid >> 7;  // 0/1
#pragma unroll
      for (int i = 0; i < 8; ++i) {
        const int kp = kpb + 2 * i;
        const bf16 lo = Bt[(size_t)(k0 + 2 * kp) * ldb + n0 + n];
        const bf16 hi = Bt[(size_t)(k0 + 2 * kp + 1) * ldb + n0 + n];
        ldsB[n * 20 + kp] = packh(lo, hi);
      }
    }
    if (k0 + 32 < K) {  // prefetch next K tile (global_prefetch_b8 path)
      __builtin_prefetch(A + (size_t)(m0 + (tid & 127)) * lda + k0 + 32, 0, 3);
      __builtin_prefetch(Bt + (size_t)(k0 + 32 + (tid >> 4)) * ldb + n0 + 8 * (tid & 15),
                         0, 3);
    }
    __syncthreads();
    const v16bf a0 = frag_a_lds(ldsA + (32 * wr) * 20, 20);
    const v16bf a1 = frag_a_lds(ldsA + (32 * wr + 16) * 20, 20);
#pragma unroll
    for (int t = 0; t < 4; ++t) {
      const v16bf bb = frag_b_lds(ldsB, 20, 64 * wc + 16 * t);
      acc[0][t] = wmma_bf16(a0, bb, acc[0][t]);
      acc[1][t] = wmma_bf16(a1, bb, acc[1][t]);
    }
  }
  const int h = lane >> 4, nl = lane & 15;
#pragma unroll
  for (int i = 0; i < 2; ++i)
#pragma unroll
    for (int t = 0; t < 4; ++t) {
      const int n = n0 + 64 * wc + 16 * t + nl;
      const float bv = bias ? bias[n] : 0.f;
#pragma unroll
      for (int r = 0; r < 8; ++r) {
        const int m = m0 + 32 * wr + 16 * i + 8 * h + r;
        float v = acc[i][t][r] + bv;
        if (relu) v = fmaxf(v, 0.f);
        if (res) v += res[(size_t)m * ldo + n];
        if (outF) outF[(size_t)m * ldo + n] = v;
        else outH[(size_t)m * ldo + n] = (bf16)v;
      }
    }
}

// d[b,c] = t[(b,c),:256] . d2_w + d2_b    (one wave per row)
__global__ void dvec_kernel(const float* __restrict__ t, const float* __restrict__ d2w,
                            const float* __restrict__ d2b, float* __restrict__ dvec) {
  const int wid = (blockIdx.x * blockDim.x + threadIdx.x) >> 5;
  const int lane = threadIdx.x & 31;
  const float* tr = t + (size_t)wid * 256;
  float s = 0.f;
#pragma unroll
  for (int i = 0; i < 8; ++i) s += tr[lane + 32 * i] * d2w[lane + 32 * i];
#pragma unroll
  for (int k = 1; k <= 16; k <<= 1) s += __shfl_xor(s, k, 32);
  if (lane == 0) dvec[wid] = s + d2b[0];
}

// per-b head: out[b] = | sum_g relu(d . c1_w[g] + c1_b[g]) * c2_w[g] + c2_b |
__global__ void head_kernel(const float* __restrict__ dvec, const float* __restrict__ c1w,
                            const float* __restrict__ c1b, const float* __restrict__ c2w,
                            const float* __restrict__ c2b, float* __restrict__ out) {
  const int b = blockIdx.x, g = threadIdx.x;  // 32 threads
  float s = 0.f;
  for (int c = 0; c < 64; ++c) s += dvec[b * 64 + c] * c1w[g * 64 + c];
  float u = fmaxf(0.f, s + c1b[g]) * c2w[g];
#pragma unroll
  for (int k = 1; k <= 16; k <<= 1) u += __shfl_xor(u, k, 32);
  if (g == 0) out[b] = fabsf(u + c2b[0]);
}

// ---------------- launch ----------------

extern "C" void kernel_launch(void* const* d_in, const int* in_sizes, int n_in,
                              void* d_out, int out_size, void* d_ws, size_t ws_size,
                              hipStream_t stream) {
  (void)in_sizes; (void)n_in; (void)out_size; (void)ws_size;
  const float* x      = (const float*)d_in[0];
  const float* occ    = (const float*)d_in[1];
  const float* proj_w = (const float*)d_in[2];
  const float* proj_b = (const float*)d_in[3];
  const float* ll1_w  = (const float*)d_in[4];
  const float* ll1_b  = (const float*)d_in[5];
  const float* ll2_w  = (const float*)d_in[6];
  const float* ll2_b  = (const float*)d_in[7];
  const float* g1_w   = (const float*)d_in[8];
  const float* g1_b   = (const float*)d_in[9];
  const float* g2_w   = (const float*)d_in[10];
  const float* g2_b   = (const float*)d_in[11];
  const float* gc_w   = (const float*)d_in[12];
  const float* gc_b   = (const float*)d_in[13];
  const float* ta_w   = (const float*)d_in[14];
  const float* ta_b   = (const float*)d_in[15];
  const float* d1_w   = (const float*)d_in[16];
  const float* d1_b   = (const float*)d_in[17];
  const float* d2_w   = (const float*)d_in[18];
  const float* d2_b   = (const float*)d_in[19];
  const float* c1_w   = (const float*)d_in[20];
  const float* c1_b   = (const float*)d_in[21];
  const float* c2_w   = (const float*)d_in[22];
  const float* c2_b   = (const float*)d_in[23];

  char* ws = (char*)d_ws;
  size_t off = 0;
  auto alloc = [&](size_t bytes) -> void* {
    void* p = (void*)(ws + off);
    off += (bytes + 255) & ~(size_t)255;
    return p;
  };
  const size_t M = 16384;  // B*C
  float* xm    = (float*)alloc(M * 512 * 4);
  bf16*  Y     = (bf16*)alloc(M * 1536 * 2);
  float* E1    = (float*)alloc(M * 64 * 4);
  float* E2    = (float*)alloc(M * 64 * 4);
  float* m1buf = (float*)alloc(M * 4);
  float* m2buf = (float*)alloc(M * 4);
  float* projB = (float*)alloc(256 * 64 * 4);
  bf16*  adjT  = (bf16*)alloc(256 * 4096 * 2);
  bf16*  llT1  = (bf16*)alloc(3 * 512 * 64 * 2);
  bf16*  llT2  = (bf16*)alloc(3 * 512 * 64 * 2);
  bf16*  gcT   = (bf16*)alloc(3 * 1536 * 512 * 2);
  bf16*  taT   = (bf16*)alloc(512 * 512 * 2);
  bf16*  d1T   = (bf16*)alloc(512 * 256 * 2);
  bf16*  zin   = (bf16*)alloc(M * 512 * 2);
  float* dvec  = (float*)alloc(M * 4);
  // tail aliases the (then-dead) Y region
  bf16*  zbuf  = (bf16*)Y;
  float* tbuf  = (float*)((char*)Y + M * 512 * 2);

  // weight prep (transpose + bf16)
  transcvt_kernel<<<384, 256, 0, stream>>>(ll1_w, llT1, 3, 64, 512);
  transcvt_kernel<<<384, 256, 0, stream>>>(ll2_w, llT2, 3, 64, 512);
  transcvt_kernel<<<9216, 256, 0, stream>>>(gc_w, gcT, 3, 512, 1536);
  transcvt_kernel<<<1024, 256, 0, stream>>>(ta_w, taT, 1, 512, 512);
  transcvt_kernel<<<512, 256, 0, stream>>>(d1_w, d1T, 1, 256, 512);

  proj_kernel<<<256, 64, 0, stream>>>(occ, proj_w, proj_b, projB);
  xpose_kernel<<<256, 256, 0, stream>>>(x, xm);

  for (int l = 0; l < 3; ++l) {
    mgate_kernel<<<2048, 256, 0, stream>>>(xm, projB, g1_w + l * 576, g2_w + l * 576,
                                           g1_b + l, g2_b + l, m1buf, m2buf);
    llsm_kernel<<<256, 128, 0, stream>>>(xm, llT1 + (size_t)l * 512 * 64,
                                         ll1_b + l * 64, m1buf, projB, E1, 0);
    llsm_kernel<<<256, 128, 0, stream>>>(xm, llT2 + (size_t)l * 512 * 64,
                                         ll2_b + l * 64, m2buf, projB, E2, 1);
    adj_kernel<<<256, 128, 0, stream>>>(E1, E2, adjT);
    nconv_kernel<<<dim3(256, 4), 256, 0, stream>>>(adjT, xm, Y, 1, 0, 1, 1);  // X1, mirror seg0
    nconv_kernel<<<dim3(256, 4), 256, 0, stream>>>(adjT, xm, Y, 0, 1, 2, 0);  // X2
    gemm_kernel<<<dim3(128, 4), 256, 0, stream>>>(
        Y, 1536, gcT + (size_t)l * 1536 * 512, 512, gc_b + l * 512, xm, nullptr,
        (l > 0) ? xm : nullptr, 512, 1536, 1);
  }

  // tail: z = xm @ taT + b ; t = relu(z @ d1T + b) ; d = t.d2 ; head
  cvt_kernel<<<8192, 256, 0, stream>>>(xm, zin, (int)(M * 512));
  gemm_kernel<<<dim3(128, 4), 256, 0, stream>>>(zin, 512, taT, 512, ta_b, nullptr, zbuf,
                                                nullptr, 512, 512, 0);
  gemm_kernel<<<dim3(128, 2), 256, 0, stream>>>(zbuf, 512, d1T, 256, d1_b, tbuf, nullptr,
                                                nullptr, 256, 512, 1);
  dvec_kernel<<<2048, 256, 0, stream>>>(tbuf, d2_w, d2_b, dvec);
  head_kernel<<<256, 32, 0, stream>>>(dvec, c1_w, c1_b, c2_w, c2_b, (float*)d_out);
}
